// SparseAttention_32461362823706
// MI455X (gfx1250) — matrix operand order repaired and checked by reading, not compiled, not run on hardware
//
#include <hip/hip_runtime.h>
#include <hip/hip_bf16.h>
#include <math.h>

// ---------------------------------------------------------------------------
// CDNA5 (gfx1250) wave32 WMMA sparse-attention block.
//  - All GEMMs: v_wmma_f32_16x16x32_bf16, NT form (B staged K-contiguous).
//  - Global->LDS staging: global_load_async_to_lds_b128 (ASYNCcnt),
//    double-buffered, overlapped with WMMA compute.
//  - Block tile 128x128x32, wave tile 32x64 (2x4 WMMA) for LDS-read economy.
// ---------------------------------------------------------------------------

typedef __attribute__((ext_vector_type(16))) __bf16         v16bf;
typedef __attribute__((ext_vector_type(8)))  float          v8f;
typedef __attribute__((ext_vector_type(8)))  unsigned short v8u;

struct FragU { v8u lo; v8u hi; };   // 32 bytes == v16bf

__device__ __forceinline__ unsigned short f32_to_bf16_rne(float f) {
  unsigned int u = __builtin_bit_cast(unsigned int, f);
  u += 0x7FFFu + ((u >> 16) & 1u);          // round-to-nearest-even
  return (unsigned short)(u >> 16);
}

// Issue one 16-byte async global->LDS copy (gfx1250, tracked by ASYNCcnt).
__device__ __forceinline__ void async_copy_b128(unsigned lds_byte_addr,
                                                const void* gaddr) {
  asm volatile("global_load_async_to_lds_b128 %0, %1, off"
               :: "v"(lds_byte_addr), "v"(gaddr) : "memory");
}

// ---------------------------------------------------------------------------
// Elementwise f32 -> bf16
// ---------------------------------------------------------------------------
__global__ __launch_bounds__(256)
void convert_f32_bf16(const float* __restrict__ in,
                      unsigned short* __restrict__ out, long n) {
  long i = (long)blockIdx.x * 256 + threadIdx.x;
  if (i < n) out[i] = f32_to_bf16_rne(in[i]);
}

// ---------------------------------------------------------------------------
// Tiled transpose + convert: out[c][r] = bf16(in[r][c]).  32x32 LDS tiles.
// ---------------------------------------------------------------------------
__global__ __launch_bounds__(256)
void transpose_convert_f32_bf16(const float* __restrict__ in,
                                unsigned short* __restrict__ out,
                                int rows, int cols) {
  __shared__ float t[32][33];
  const int tx = threadIdx.x & 31;
  const int ty = threadIdx.x >> 5;
  const int r0 = blockIdx.y * 32;
  const int c0 = blockIdx.x * 32;
#pragma unroll
  for (int i = 0; i < 4; ++i)
    t[ty + 8 * i][tx] = in[(long)(r0 + ty + 8 * i) * cols + c0 + tx];
  __syncthreads();
#pragma unroll
  for (int i = 0; i < 4; ++i)
    out[(long)(c0 + ty + 8 * i) * rows + r0 + tx] =
        f32_to_bf16_rne(t[tx][ty + 8 * i]);
}

// ---------------------------------------------------------------------------
// NT WMMA GEMM:  C[m][n] = sum_k A[m][k] * B[n][k]     (both K-contiguous)
// EPI: 0 = bf16 row-major store
//      1 = f32  row-major store
//      2 = scores: scale + hard threshold, f32 store
//      3 = bf16 store transposed per batch of 2048 rows: vt[b][n][s]
// Block tile 128x128x32; 8 waves (4 down M x 2 across N); wave tile 32x64
// (2x4 WMMA).  Async double-buffered LDS staging.
// ---------------------------------------------------------------------------
template<int EPI>
__global__ __launch_bounds__(256)
void gemm_wmma_nt(const unsigned short* __restrict__ A,
                  const unsigned short* __restrict__ B,
                  void* __restrict__ Cv,
                  int K, int lda, int ldb, int ldc,
                  long batchStrideA, long batchStrideB, long batchStrideC,
                  float scale, float thresh)
{
  constexpr int BM = 128, BN = 128, BK = 32;
  __shared__ unsigned short sA[2][BM * BK];   // [m][k], 8 KB per buffer
  __shared__ unsigned short sB[2][BN * BK];   // [n][k], 8 KB per buffer

  const int tid   = threadIdx.x;
  const int lane  = tid & 31;
  const int wave  = tid >> 5;
  const int waveM = wave & 3;                 // 4 waves down M (32 rows each)
  const int waveN = wave >> 2;                // 2 waves across N (64 cols each)

  const long b = blockIdx.z;
  A += b * batchStrideA;
  B += b * batchStrideB;

  const int m0 = blockIdx.y * BM;
  const int n0 = blockIdx.x * BN;

  // Per-thread staging coordinates: 32 bytes of A and 32 bytes of B.
  const int ar = tid >> 1;                    // row 0..127
  const int ac = (tid & 1) * 16;              // col 0 or 16

  // Generic->LDS pointers: low 32 bits of a generic LDS pointer are the
  // LDS byte offset (flat-aperture rule), which is what the async op wants.
  const unsigned sA0 = (unsigned)(size_t)&sA[0][0];
  const unsigned sB0 = (unsigned)(size_t)&sB[0][0];
  const unsigned ldsA = sA0 + (unsigned)(ar * BK + ac) * 2u;
  const unsigned ldsB = sB0 + (unsigned)(ar * BK + ac) * 2u;
  constexpr unsigned bufA = BM * BK * 2u;     // bytes per A buffer
  constexpr unsigned bufB = BN * BK * 2u;     // bytes per B buffer

  const int nK = K / BK;

  auto stage = [&](int buf, int kt) {
    const int k0 = kt * BK;
    const unsigned short* ga = A + (long)(m0 + ar) * lda + k0 + ac;
    const unsigned short* gb = B + (long)(n0 + ar) * ldb + k0 + ac;
    async_copy_b128(ldsA + buf * bufA,       ga);
    async_copy_b128(ldsA + buf * bufA + 16u, ga + 8);
    async_copy_b128(ldsB + buf * bufB,       gb);
    async_copy_b128(ldsB + buf * bufB + 16u, gb + 8);
  };

  v8f acc[2][4] = {};
  stage(0, 0);                                // prologue: tile 0 in flight

  for (int kt = 0; kt < nK; ++kt) {
    const int cur = kt & 1;
    if (kt + 1 < nK) {
      stage(1 - cur, kt + 1);                 // overlap next tile's copies
      asm volatile("s_wait_asynccnt 0x4" ::: "memory");  // cur tile landed
    } else {
      asm volatile("s_wait_asynccnt 0x0" ::: "memory");
    }
    __syncthreads();                          // LDS visible to all waves

    const int hlf = lane >> 4;                // per ISA 7.12.2 layouts
    const int lm  = lane & 15;

    // A fragments (16x32 bf16)
    v16bf av[2];
#pragma unroll
    for (int mt = 0; mt < 2; ++mt) {
      const int row = waveM * 32 + mt * 16 + lm;
      FragU f;
      f.lo = *(const v8u*)&sA[cur][row * BK + hlf * 8];
      f.hi = *(const v8u*)&sA[cur][row * BK + 16 + hlf * 8];
      av[mt] = __builtin_bit_cast(v16bf, f);
    }
    // B fragments (32x16 bf16) from the [n][k] tile
    v16bf bv[4];
#pragma unroll
    for (int nt = 0; nt < 4; ++nt) {
      const int col = waveN * 64 + nt * 16 + lm;
      FragU f;
      f.lo = *(const v8u*)&sB[cur][col * BK + hlf * 8];
      f.hi = *(const v8u*)&sB[cur][col * BK + 16 + hlf * 8];
      bv[nt] = __builtin_bit_cast(v16bf, f);
    }
#pragma unroll
    for (int mt = 0; mt < 2; ++mt)
#pragma unroll
      for (int nt = 0; nt < 4; ++nt)
        acc[mt][nt] = __builtin_amdgcn_wmma_f32_16x16x32_bf16(
            false, av[mt], false, bv[nt], (short)0, acc[mt][nt], false, false);

    __syncthreads();                          // done reading buf[cur]
  }

  // ---- epilogue: C layout: VGPR r -> M = r + 8*half, N = lane%16
  const int hlf = lane >> 4;
  const int lm  = lane & 15;
#pragma unroll
  for (int mt = 0; mt < 2; ++mt) {
#pragma unroll
    for (int nt = 0; nt < 4; ++nt) {
      const int colg  = n0 + waveN * 64 + nt * 16 + lm;
      const int rowg0 = m0 + waveM * 32 + mt * 16 + hlf * 8;
      if (EPI == 3) {
        // transposed bf16 store: 8 consecutive rows -> one b128
        v8u pk;
#pragma unroll
        for (int r = 0; r < 8; ++r) pk[r] = f32_to_bf16_rne(acc[mt][nt][r]);
        const long bb = (long)(rowg0 >> 11);         // batch of 2048 rows
        unsigned short* dst = (unsigned short*)Cv +
            bb * batchStrideC + (long)colg * ldc + (rowg0 & 2047);
        *(v8u*)dst = pk;
      } else {
#pragma unroll
        for (int r = 0; r < 8; ++r) {
          float val = acc[mt][nt][r];
          if (EPI == 2) { val *= scale; if (val < thresh) val = 0.0f; }
          const long idx = b * batchStrideC + (long)(rowg0 + r) * ldc + colg;
          if (EPI == 0) ((unsigned short*)Cv)[idx] = f32_to_bf16_rne(val);
          else          ((float*)Cv)[idx] = val;
        }
      }
    }
  }
}

// ---------------------------------------------------------------------------
// Row softmax over S=2048.  Final attn f32 in place + bf16 copy.
// ---------------------------------------------------------------------------
__global__ __launch_bounds__(256)
void softmax_rows(float* __restrict__ att, unsigned short* __restrict__ att_bf)
{
  __shared__ float red[256];
  const long row = blockIdx.x;
  const int  tid = threadIdx.x;
  float* p = att + row * 2048;

  float x[8];
  float mx = -INFINITY;
#pragma unroll
  for (int i = 0; i < 8; ++i) { x[i] = p[tid + i * 256]; mx = fmaxf(mx, x[i]); }
  red[tid] = mx; __syncthreads();
  for (int s = 128; s > 0; s >>= 1) {
    if (tid < s) red[tid] = fmaxf(red[tid], red[tid + s]);
    __syncthreads();
  }
  mx = red[0]; __syncthreads();

  float sum = 0.f;
#pragma unroll
  for (int i = 0; i < 8; ++i) { x[i] = __expf(x[i] - mx); sum += x[i]; }
  red[tid] = sum; __syncthreads();
  for (int s = 128; s > 0; s >>= 1) {
    if (tid < s) red[tid] += red[tid + s];
    __syncthreads();
  }
  const float inv = 1.0f / red[0];

#pragma unroll
  for (int i = 0; i < 8; ++i) {
    const float v = x[i] * inv;
    p[tid + i * 256] = v;
    att_bf[row * 2048 + tid + i * 256] = f32_to_bf16_rne(v);
  }
}

// ---------------------------------------------------------------------------
// out = LayerNorm(t + residual) * gamma + beta   (row = 1024)
// ---------------------------------------------------------------------------
__global__ __launch_bounds__(256)
void residual_layernorm(const float* __restrict__ t, const float* __restrict__ res,
                        const float* __restrict__ gamma, const float* __restrict__ beta,
                        float* __restrict__ out)
{
  __shared__ float red[256];
  const long row = blockIdx.x;
  const int  tid = threadIdx.x;

  float x[4]; float s = 0.f;
#pragma unroll
  for (int i = 0; i < 4; ++i) {
    const int c = tid + i * 256;
    x[i] = t[row * 1024 + c] + res[row * 1024 + c];
    s += x[i];
  }
  red[tid] = s; __syncthreads();
  for (int st = 128; st > 0; st >>= 1) {
    if (tid < st) red[tid] += red[tid + st];
    __syncthreads();
  }
  const float mean = red[0] * (1.0f / 1024.0f);
  __syncthreads();

  float vs = 0.f;
#pragma unroll
  for (int i = 0; i < 4; ++i) { const float d = x[i] - mean; vs += d * d; }
  red[tid] = vs; __syncthreads();
  for (int st = 128; st > 0; st >>= 1) {
    if (tid < st) red[tid] += red[tid + st];
    __syncthreads();
  }
  const float rstd = rsqrtf(red[0] * (1.0f / 1024.0f) + 1e-6f);

#pragma unroll
  for (int i = 0; i < 4; ++i) {
    const int c = tid + i * 256;
    out[row * 1024 + c] = (x[i] - mean) * rstd * gamma[c] + beta[c];
  }
}

// ---------------------------------------------------------------------------
extern "C" void kernel_launch(void* const* d_in, const int* in_sizes, int n_in,
                              void* d_out, int out_size, void* d_ws, size_t ws_size,
                              hipStream_t stream)
{
  (void)in_sizes; (void)n_in; (void)out_size; (void)ws_size;

  const float* q   = (const float*)d_in[0];
  const float* k   = (const float*)d_in[1];
  const float* v   = (const float*)d_in[2];
  const float* Wq  = (const float*)d_in[3];
  const float* Wk  = (const float*)d_in[4];
  const float* Wv  = (const float*)d_in[5];
  const float* Wfc = (const float*)d_in[6];
  const float* ga  = (const float*)d_in[7];
  const float* be  = (const float*)d_in[8];

  const int  Bb = 8, S = 2048, D = 1024;
  const long SD = (long)Bb * S * D;     // 16,777,216
  const long SS = (long)Bb * S * S;     // 33,554,432
  const long W  = (long)D * D;          // 1,048,576

  float* out_main = (float*)d_out;       // [B,S,D]
  float* out_att  = (float*)d_out + SD;  // [B,S,S]

  // workspace carve-up (ushort units)
  unsigned short* ws    = (unsigned short*)d_ws;
  unsigned short* wq_t  = ws;            // W^T (bf16), all GEMMs are NT
  unsigned short* wk_t  = wq_t + W;
  unsigned short* wv_t  = wk_t + W;
  unsigned short* wfc_t = wv_t + W;
  unsigned short* qh    = wfc_t + W;     // [B*S][D] bf16
  unsigned short* kh    = qh + SD;       // [B*S][D] bf16
  unsigned short* vh_t  = kh + SD;       // [B][D][S] bf16 (pre-transposed)
  unsigned short* s0    = vh_t + SD;     // staging (q/k/v bf16), later ctx bf16
  unsigned short* attb  = s0 + SD;       // attn bf16 [B][S][S]
  float*          f32t  = (float*)(attb + SS);  // pre-LN f32

  const dim3 blk(256);
  auto cgrid = [](long n) { return dim3((unsigned)((n + 255) / 256)); };
  const float inv_sqrt_dk = 0.03125f;    // 1/sqrt(1024)

  // ---- weights -> bf16, transposed (NT B operand)
  const dim3 tgrid(D / 32, D / 32);
  transpose_convert_f32_bf16<<<tgrid, blk, 0, stream>>>(Wq,  wq_t,  D, D);
  transpose_convert_f32_bf16<<<tgrid, blk, 0, stream>>>(Wk,  wk_t,  D, D);
  transpose_convert_f32_bf16<<<tgrid, blk, 0, stream>>>(Wv,  wv_t,  D, D);
  transpose_convert_f32_bf16<<<tgrid, blk, 0, stream>>>(Wfc, wfc_t, D, D);

  // ---- projections: [16384,1024] x [1024,1024]^T
  convert_f32_bf16<<<cgrid(SD), blk, 0, stream>>>(q, s0, SD);
  gemm_wmma_nt<0><<<dim3(D / 128, (Bb * S) / 128, 1), blk, 0, stream>>>(
      s0, wq_t, qh, D, D, D, D, 0, 0, 0, 1.f, 0.f);
  convert_f32_bf16<<<cgrid(SD), blk, 0, stream>>>(k, s0, SD);
  gemm_wmma_nt<0><<<dim3(D / 128, (Bb * S) / 128, 1), blk, 0, stream>>>(
      s0, wk_t, kh, D, D, D, D, 0, 0, 0, 1.f, 0.f);
  // v projection stores vh transposed per batch: vh_t[b][d][s]
  convert_f32_bf16<<<cgrid(SD), blk, 0, stream>>>(v, s0, SD);
  gemm_wmma_nt<3><<<dim3(D / 128, (Bb * S) / 128, 1), blk, 0, stream>>>(
      s0, wv_t, vh_t, D, D, D, /*ldc=*/S, 0, 0, /*batchStrideC=*/(long)D * S,
      1.f, 0.f);

  // ---- scores = qh * kh^T * inv_sqrt_dk, hard threshold; f32 into attn out
  gemm_wmma_nt<2><<<dim3(S / 128, S / 128, Bb), blk, 0, stream>>>(
      qh, kh, out_att, D, D, D, S,
      (long)S * D, (long)S * D, (long)S * S, inv_sqrt_dk, 0.9f);

  // ---- softmax rows (final attn f32 + bf16 copy)
  softmax_rows<<<dim3(Bb * S), blk, 0, stream>>>(out_att, attb);

  // ---- ctx = attn * vh  == NT against vh_t[b][d][s]  -> s0 (bf16 [B*S][D])
  gemm_wmma_nt<0><<<dim3(D / 128, S / 128, Bb), blk, 0, stream>>>(
      attb, vh_t, s0, S, S, S, D,
      (long)S * S, (long)D * S, (long)S * D, 1.f, 0.f);

  // ---- f32t = ctx * Wfc^T(NT)
  gemm_wmma_nt<1><<<dim3(D / 128, (Bb * S) / 128, 1), blk, 0, stream>>>(
      s0, wfc_t, f32t, D, D, D, D, 0, 0, 0, 1.f, 0.f);

  // ---- out = LayerNorm(f32t + q)
  residual_layernorm<<<dim3(Bb * S), blk, 0, stream>>>(f32t, q, ga, be, out_main);
}
